// TransitionDown_420906795556
// MI455X (gfx1250) — compile-verified, matrix operand order checked by reading, not run on hardware
//
#include <hip/hip_runtime.h>
#include <hip/hip_bf16.h>

#define NPTS      16384
#define IN_PLANES 64
#define OUT_PLANES 128
#define STRIDE_   4
#define NSAMPLE   16
#define MPTS      (NPTS / STRIDE_)      // 4096
#define ROWS      (MPTS * NSAMPLE)      // 65536
#define KPAD      128
#define EPSV      1e-5f

typedef __attribute__((ext_vector_type(16))) _Float16 v16h;
typedef __attribute__((ext_vector_type(8)))  _Float16 half8;
typedef __attribute__((ext_vector_type(8)))  float    v8f;

union V16 { v16h v; half8 h[2]; };

// ---------------- workspace layout (bytes) ----------------
// dist   @ 0        : NPTS*4      = 65536
// fpsidx @ 65536    : MPTS*4      = 16384
// knn    @ 81920    : MPTS*16*4   = 262144
// sums   @ 344064   : 128*4
// sumsq  @ 344576   : 128*4
// y      @ 345088   : ROWS*128*4  = 33554432
// total ~= 33.9 MB

// ---------------- init: zero BN accumulators, write n_o ----------------
__global__ __launch_bounds__(64)
void k_init(const int* __restrict__ o, float* __restrict__ sums,
            float* __restrict__ sumsq, float* __restrict__ out) {
  int t = threadIdx.x;
  for (int i = t; i < OUT_PLANES; i += 64) { sums[i] = 0.f; sumsq[i] = 0.f; }
  if (t == 0) {
    ((int*)out)[3 * MPTS + OUT_PLANES * MPTS] = o[0] / STRIDE_;
  }
}

// ---------------- furthest point sampling (serial chain) ----------------
__global__ __launch_bounds__(1024)
void k_fps(const float* __restrict__ p, float* __restrict__ dist,
           int* __restrict__ fpsidx) {
  const int tid = threadIdx.x;
  __shared__ float rv[32];
  __shared__ int   ri[32];
  __shared__ int   s_last;

  for (int i = tid; i < NPTS; i += 1024) dist[i] = 1e10f;
  if (tid == 0) { fpsidx[0] = 0; s_last = 0; }
  __syncthreads();

  for (int step = 1; step < MPTS; ++step) {
    const int last = s_last;
    const float lx = p[last * 3 + 0];
    const float ly = p[last * 3 + 1];
    const float lz = p[last * 3 + 2];
    float bv = -1.f; int bj = 0x7fffffff;
    for (int j = tid; j < NPTS; j += 1024) {
      float dx = p[j * 3 + 0] - lx;
      float dy = p[j * 3 + 1] - ly;
      float dz = p[j * 3 + 2] - lz;
      float d  = dx * dx + dy * dy + dz * dz;
      float dj = fminf(dist[j], d);
      dist[j] = dj;
      if (dj > bv) { bv = dj; bj = j; }   // strict > keeps earliest index
    }
    // wave32 argmax reduce (tie -> lower index)
    #pragma unroll
    for (int off = 16; off > 0; off >>= 1) {
      float ov = __shfl_down(bv, off, 32);
      int   oj = __shfl_down(bj, off, 32);
      if (ov > bv || (ov == bv && oj < bj)) { bv = ov; bj = oj; }
    }
    if ((tid & 31) == 0) { rv[tid >> 5] = bv; ri[tid >> 5] = bj; }
    __syncthreads();
    if (tid < 32) {
      bv = rv[tid]; bj = ri[tid];
      #pragma unroll
      for (int off = 16; off > 0; off >>= 1) {
        float ov = __shfl_down(bv, off, 32);
        int   oj = __shfl_down(bj, off, 32);
        if (ov > bv || (ov == bv && oj < bj)) { bv = ov; bj = oj; }
      }
      if (tid == 0) { s_last = bj; fpsidx[step] = bj; }
    }
    __syncthreads();
  }
}

// ---------------- gather sampled coordinates n_p ----------------
__global__ __launch_bounds__(256)
void k_np(const float* __restrict__ p, const int* __restrict__ fpsidx,
          float* __restrict__ out) {
  int t = blockIdx.x * blockDim.x + threadIdx.x;
  if (t < MPTS) {
    int idx = fpsidx[t];
    out[t * 3 + 0] = p[idx * 3 + 0];
    out[t * 3 + 1] = p[idx * 3 + 1];
    out[t * 3 + 2] = p[idx * 3 + 2];
  }
}

// ---------------- kNN: one block per query ----------------
__global__ __launch_bounds__(128)
void k_knn(const float* __restrict__ p, const float* __restrict__ npnts,
           int* __restrict__ knn) {
  const int m   = blockIdx.x;
  const int tid = threadIdx.x;
  const float qx = npnts[m * 3 + 0];
  const float qy = npnts[m * 3 + 1];
  const float qz = npnts[m * 3 + 2];

  float bd[NSAMPLE]; int bi[NSAMPLE];
  #pragma unroll
  for (int t = 0; t < NSAMPLE; ++t) { bd[t] = 3.4e38f; bi[t] = 0x7fffffff; }

  for (int j = tid; j < NPTS; j += 128) {
    float dx = p[j * 3 + 0] - qx;
    float dy = p[j * 3 + 1] - qy;
    float dz = p[j * 3 + 2] - qz;
    float d  = dx * dx + dy * dy + dz * dz;
    if (d < bd[NSAMPLE - 1]) {
      bd[NSAMPLE - 1] = d; bi[NSAMPLE - 1] = j;
      #pragma unroll
      for (int t = NSAMPLE - 1; t > 0; --t) {       // bubble up, static indices
        if (bd[t] < bd[t - 1]) {
          float td = bd[t]; bd[t] = bd[t - 1]; bd[t - 1] = td;
          int   ti = bi[t]; bi[t] = bi[t - 1]; bi[t - 1] = ti;
        }
      }
    }
  }

  __shared__ float sd[128 * NSAMPLE];
  __shared__ int   si[128 * NSAMPLE];
  __shared__ float rv[128];
  __shared__ int   rj[128];
  __shared__ int   rp[128];

  #pragma unroll
  for (int t = 0; t < NSAMPLE; ++t) { sd[tid * NSAMPLE + t] = bd[t]; si[tid * NSAMPLE + t] = bi[t]; }
  __syncthreads();

  for (int r = 0; r < NSAMPLE; ++r) {
    float mv = 3.4e38f; int mj = 0x7fffffff; int mp = 0;
    for (int t = tid; t < 128 * NSAMPLE; t += 128) {
      float v = sd[t]; int j = si[t];
      if (v < mv || (v == mv && j < mj)) { mv = v; mj = j; mp = t; }
    }
    rv[tid] = mv; rj[tid] = mj; rp[tid] = mp;
    __syncthreads();
    for (int s = 64; s > 0; s >>= 1) {
      if (tid < s) {
        float ov = rv[tid + s]; int oj = rj[tid + s];
        if (ov < rv[tid] || (ov == rv[tid] && oj < rj[tid])) {
          rv[tid] = ov; rj[tid] = oj; rp[tid] = rp[tid + s];
        }
      }
      __syncthreads();
    }
    if (tid == 0) {
      knn[m * NSAMPLE + r] = rj[0];
      sd[rp[0]] = 3.4e38f;
    }
    __syncthreads();
  }
}

// ---------------- fused feature build + WMMA GEMM ----------------
// A: (16 samples of query m) x KPAD  f16 in LDS, per wave
// B: W as f16 rows (out channel x K) in LDS, shared by block  (B = W^T => B cols == W rows)
__global__ __launch_bounds__(256)
void k_gemm(const float* __restrict__ p, const float* __restrict__ x,
            const float* __restrict__ W, const int* __restrict__ knn,
            const float* __restrict__ npnts, float* __restrict__ y) {
  __shared__ _Float16 ldsW[OUT_PLANES * KPAD];   // 32 KB
  __shared__ _Float16 ldsA[8 * 16 * KPAD];       // 32 KB (8 waves x 16x128 tile)

  const int tid  = threadIdx.x;
  const int lane = tid & 31;
  const int w    = tid >> 5;
  const int m    = blockIdx.x * 8 + w;           // 512 blocks * 8 waves = 4096 queries

  // stage W -> f16, zero pad K 67..127
  for (int i = tid; i < OUT_PLANES * KPAD; i += 256) {
    int r = i >> 7, c = i & (KPAD - 1);
    ldsW[i] = (c < 67) ? (_Float16)W[r * 67 + c] : (_Float16)0.0f;
  }

  // stage A tile for this wave: [rel_xyz(3) | x(64) | pad(61)]
  _Float16* aw = &ldsA[w * 16 * KPAD];
  const float npx = npnts[m * 3 + 0];
  const float npy = npnts[m * 3 + 1];
  const float npz = npnts[m * 3 + 2];
  for (int s = 0; s < NSAMPLE; ++s) {
    int kj = knn[m * NSAMPLE + s];
    float x0 = x[kj * IN_PLANES + 2 * lane + 0];
    float x1 = x[kj * IN_PLANES + 2 * lane + 1];
    aw[s * KPAD + 3 + 2 * lane] = (_Float16)x0;
    aw[s * KPAD + 4 + 2 * lane] = (_Float16)x1;
    if (lane < 3) {
      float pc = p[kj * 3 + lane];
      float nc = (lane == 0) ? npx : ((lane == 1) ? npy : npz);
      aw[s * KPAD + lane] = (_Float16)(pc - nc);
    }
    int c = 67 + lane;
    aw[s * KPAD + c] = (_Float16)0.0f;             // 67..98
    c = 99 + lane;
    if (c < KPAD) aw[s * KPAD + c] = (_Float16)0.0f; // 99..127
  }
  __syncthreads();

  // per-lane fragment addressing per CDNA5 16-bit A layout:
  // lanes<16: K {0..7,16..23}; lanes>=16: K {8..15,24..31} of each 32-chunk
  const int hoff = (lane < 16) ? 0 : 8;
  const int arow = lane & 15;

  V16 afrag[4];
  #pragma unroll
  for (int k = 0; k < 4; ++k) {
    const _Float16* base = &aw[arow * KPAD + 32 * k + hoff];
    afrag[k].h[0] = *(const half8*)(base);
    afrag[k].h[1] = *(const half8*)(base + 16);
  }

  #pragma unroll
  for (int nt = 0; nt < 8; ++nt) {
    v8f acc = {};
    #pragma unroll
    for (int k = 0; k < 4; ++k) {
      V16 bfrag;
      const _Float16* bb = &ldsW[(nt * 16 + arow) * KPAD + 32 * k + hoff];
      bfrag.h[0] = *(const half8*)(bb);
      bfrag.h[1] = *(const half8*)(bb + 16);
      acc = __builtin_amdgcn_wmma_f32_16x16x32_f16(
          /*neg_a=*/false, afrag[k].v, /*neg_b=*/false, bfrag.v,
          /*c_mod=*/(short)0, acc, /*reuse_a=*/false, /*reuse_b=*/false);
    }
    // C/D layout: VGPR v -> M = v (lanes<16) or v+8 (lanes>=16); N = lane&15
    #pragma unroll
    for (int v = 0; v < 8; ++v) {
      int row = m * 16 + v + ((lane < 16) ? 0 : 8);
      int col = nt * 16 + arow;
      y[row * OUT_PLANES + col] = acc[v];
    }
  }
}

// ---------------- BN statistics (population mean/var per channel) ----------------
__global__ __launch_bounds__(256)
void k_stats(const float* __restrict__ y, float* __restrict__ sums,
             float* __restrict__ sumsq) {
  int t = blockIdx.x * 256 + threadIdx.x;   // 65536 threads
  int c = t & (OUT_PLANES - 1);
  int g = t >> 7;                           // 0..511
  float s = 0.f, s2 = 0.f;
  for (int r = g; r < ROWS; r += 512) {
    float v = y[r * OUT_PLANES + c];
    s += v; s2 += v * v;
  }
  atomicAdd(&sums[c], s);
  atomicAdd(&sumsq[c], s2);
}

// ---------------- normalize + ReLU + max-pool over nsample ----------------
__global__ __launch_bounds__(256)
void k_final(const float* __restrict__ y, const float* __restrict__ sums,
             const float* __restrict__ sumsq, const float* __restrict__ gamma,
             const float* __restrict__ beta, float* __restrict__ out) {
  int t = blockIdx.x * 256 + threadIdx.x;
  if (t >= MPTS * OUT_PLANES) return;
  int m = t >> 7;
  int c = t & (OUT_PLANES - 1);
  const float inv = 1.f / (float)ROWS;
  float mean  = sums[c] * inv;
  float var   = sumsq[c] * inv - mean * mean;
  float scale = gamma[c] * rsqrtf(var + EPSV);
  float b     = beta[c];
  float best  = 0.f;                        // relu output >= 0
  #pragma unroll
  for (int s = 0; s < NSAMPLE; ++s) {
    float v = y[(m * NSAMPLE + s) * OUT_PLANES + c];
    v = fmaxf((v - mean) * scale + b, 0.f);
    best = fmaxf(best, v);
  }
  out[3 * MPTS + m * OUT_PLANES + c] = best;
}

extern "C" void kernel_launch(void* const* d_in, const int* in_sizes, int n_in,
                              void* d_out, int out_size, void* d_ws, size_t ws_size,
                              hipStream_t stream) {
  const float* p     = (const float*)d_in[0];
  const float* x     = (const float*)d_in[1];
  const int*   o     = (const int*)d_in[2];
  const float* W     = (const float*)d_in[3];
  const float* gamma = (const float*)d_in[4];
  const float* beta  = (const float*)d_in[5];
  float* out = (float*)d_out;

  char* ws = (char*)d_ws;
  float* dist   = (float*)(ws + 0);
  int*   fpsidx = (int*)  (ws + 65536);
  int*   knn    = (int*)  (ws + 81920);
  float* sums   = (float*)(ws + 344064);
  float* sumsq  = (float*)(ws + 344576);
  float* y      = (float*)(ws + 345088);

  k_init <<<1, 64, 0, stream>>>(o, sums, sumsq, out);
  k_fps  <<<1, 1024, 0, stream>>>(p, dist, fpsidx);
  k_np   <<<(MPTS + 255) / 256, 256, 0, stream>>>(p, fpsidx, out);
  k_knn  <<<MPTS, 128, 0, stream>>>(p, out, knn);
  k_gemm <<<MPTS / 8, 256, 0, stream>>>(p, x, W, knn, out, y);
  k_stats<<<256, 256, 0, stream>>>(y, sums, sumsq);
  k_final<<<(MPTS * OUT_PLANES + 255) / 256, 256, 0, stream>>>(y, sums, sumsq, gamma, beta, out);
}
